// Attention_27590869910208
// MI455X (gfx1250) — matrix operand order repaired and checked by reading, not compiled, not run on hardware
//
#include <hip/hip_runtime.h>
#include <hip/hip_bf16.h>

typedef __attribute__((ext_vector_type(16))) _Float16 v16h;
typedef __attribute__((ext_vector_type(8)))  _Float16 v8h;
typedef __attribute__((ext_vector_type(8)))  float    v8f;

#define N_HEADS  16
#define KV_HEADS 4
#define HEAD_DIM 128
#define BATCH    2
#define SEQ      2048
#define HID      2048
#define KV_DIM   512   // KV_HEADS * HEAD_DIM

__device__ __forceinline__ v8f wmma16(v16h a, v16h b, v8f c) {
  // 8 args: (neg_a, A, neg_b, B, c_mod, C, reuse_a, reuse_b)
  return __builtin_amdgcn_wmma_f32_16x16x32_f16(false, a, false, b, (short)0, c,
                                                false, false);
}

__device__ __forceinline__ v16h pack16(v8h lo, v8h hi) {
  v16h r;
#pragma unroll
  for (int i = 0; i < 8; ++i) { r[i] = lo[i]; r[i + 8] = hi[i]; }
  return r;
}

// Pin 4 fragments live simultaneously so the scheduler issues all their loads
// as one clause (single counter wait) and the 4 WMMAs go back-to-back.
__device__ __forceinline__ void pin4(v16h& a, v16h& b, v16h& c, v16h& d) {
  asm volatile("" : "+v"(a), "+v"(b), "+v"(c), "+v"(d));
}

// ---------------------------------------------------------------- convert
__global__ void cvt_f32_f16(const float* __restrict__ in,
                            _Float16* __restrict__ out, size_t n) {
  size_t i = (size_t)blockIdx.x * blockDim.x + threadIdx.x;
  size_t stride = (size_t)gridDim.x * blockDim.x;
  for (; i < n; i += stride) out[i] = (_Float16)in[i];
}

// ---------------------------------------------------------------- GEMM
// C[M,N] = A[M,K] * W[N,K]^T + bias[N]
// Block tile 128(M) x 64(N): 8 waves, wave w owns rows [bm+16w, bm+16w+16).
// W tile (64x32 f16 = 4KB) staged per k-step into double-buffered LDS via
// global_load_async_to_lds_b128 (ASYNCcnt), shared by all 8 waves.
// A fragments register-resident with one-step prefetch; B fragments pinned
// into distinct registers so the 4 WMMAs issue back-to-back after one wait.
__global__ __launch_bounds__(256) void gemm_wmma(
    const _Float16* __restrict__ A, const _Float16* __restrict__ W,
    const float* __restrict__ bias, _Float16* __restrict__ outh,
    float* __restrict__ outf, int M, int N, int K) {
  __shared__ __align__(16) _Float16 wtile[2][64 * 32];
  int lane = threadIdx.x & 31;
  int wave = threadIdx.x >> 5;
  int nb = N >> 6;
  int bm = (int)(blockIdx.x / nb) * 128;
  int n0 = (int)(blockIdx.x % nb) * 64;
  int row16 = lane & 15, hi = lane >> 4;

  // staging: thread t loads W[n0 + (t>>2)][k0 + (t&3)*8 .. +7] (one b128)
  int wrow = threadIdx.x >> 2;
  int wchunk = (threadIdx.x & 3) * 8;
  const _Float16* wsrc0 = W + (size_t)(n0 + wrow) * K + wchunk;

  auto stage = [&](int k0, int buf) {
    unsigned lds = (unsigned)(size_t)(&wtile[buf][wrow * 32 + wchunk]);
    const _Float16* g = wsrc0 + k0;
    asm volatile("global_load_async_to_lds_b128 %0, %1, off"
                 :: "v"(lds), "v"(g) : "memory");
  };

  const _Float16* arow = A + (size_t)(bm + wave * 16 + row16) * K;
  auto loadA = [&](int k0) {
    // A 16x32 fragment: lanes<16 hold K 0..7 & 16..23, lanes>=16 8..15 & 24..31
    return pack16(*(const v8h*)(arow + k0 + hi * 8),
                  *(const v8h*)(arow + k0 + hi * 8 + 16));
  };

  stage(0, 0);
  v16h a_cur = loadA(0);
  asm volatile("s_wait_asynccnt 0x0" ::: "memory");
  __syncthreads();

  v8f acc[4] = {};
  int nk = K >> 5;
  for (int i = 0; i < nk; ++i) {
    int buf = i & 1;
    bool more = (i + 1) < nk;
    if (more) stage((i + 1) * 32, buf ^ 1);
    v16h a_nxt = more ? loadA((i + 1) * 32) : a_cur;
    const _Float16* bt = &wtile[buf][0];
    // stage all 4 B fragments into distinct registers, then 4 WMMAs
    v16h bfr[4];
#pragma unroll
    for (int j = 0; j < 4; ++j) {
      // B 32x16 fragment: column n = W row; lanes<16 K=0..7 & 16..23, lanes>=16 8..15 & 24..31
      const _Float16* bp = bt + (j * 16 + row16) * 32 + hi * 8;
      bfr[j] = pack16(*(const v8h*)bp, *(const v8h*)(bp + 16));
    }
    pin4(bfr[0], bfr[1], bfr[2], bfr[3]);
#pragma unroll
    for (int j = 0; j < 4; ++j) acc[j] = wmma16(a_cur, bfr[j], acc[j]);
    a_cur = a_nxt;
    if (more) {
      asm volatile("s_wait_asynccnt 0x0" ::: "memory");
      __syncthreads();
    }
  }
#pragma unroll
  for (int j = 0; j < 4; ++j) {
    int col = n0 + j * 16 + row16;
    float bv = bias[col];
#pragma unroll
    for (int r = 0; r < 8; ++r) {
      int m = bm + wave * 16 + r + hi * 8;
      float v = acc[j][r] + bv;
      if (outf) outf[(size_t)m * N + col] = v;
      else      outh[(size_t)m * N + col] = (_Float16)v;
    }
  }
}

// ---------------------------------------------------------------- RoPE (in place, Q then K)
__global__ void rope_qk(_Float16* __restrict__ q, _Float16* __restrict__ k,
                        const float* __restrict__ cs, const float* __restrict__ sn) {
  size_t i = (size_t)blockIdx.x * blockDim.x + threadIdx.x;
  size_t total = (size_t)BATCH * SEQ * (N_HEADS + KV_HEADS) * 64;
  if (i >= total) return;
  int pair = (int)(i & 63);
  size_t t = i >> 6;
  int head = (int)(t % (N_HEADS + KV_HEADS)); t /= (N_HEADS + KV_HEADS);
  int s = (int)(t % SEQ);
  int b = (int)(t / SEQ);
  _Float16* base = (head < N_HEADS)
      ? q + ((size_t)(b * SEQ + s) * N_HEADS + head) * HEAD_DIM
      : k + ((size_t)(b * SEQ + s) * KV_HEADS + (head - N_HEADS)) * HEAD_DIM;
  float c  = cs[s * HEAD_DIM + pair];
  float sv = sn[s * HEAD_DIM + pair];
  float x0 = (float)base[pair];
  float x1 = (float)base[pair + 64];
  base[pair]      = (_Float16)(x0 * c - x1 * sv);
  base[pair + 64] = (_Float16)(x1 * c + x0 * sv);
}

// ---------------------------------------------------------------- V transpose -> (b, kvh, d, s)
__global__ void v_transpose(const _Float16* __restrict__ v,
                            _Float16* __restrict__ vt) {
  size_t i = (size_t)blockIdx.x * blockDim.x + threadIdx.x;
  size_t total = (size_t)BATCH * SEQ * KV_HEADS * HEAD_DIM;
  if (i >= total) return;
  int d = (int)(i & 127);
  size_t t = i >> 7;
  int kv = (int)(t & 3); t >>= 2;
  int s = (int)(t & 2047);
  int b = (int)(t >> 11);
  vt[(((size_t)b * KV_HEADS + kv) * HEAD_DIM + d) * SEQ + s] = v[i];
}

// ---------------------------------------------------------------- flash attention
// one wave = one (b, h, 16-row q-tile); causal online softmax, 32 keys/step
__global__ __launch_bounds__(256) void flash_attn(
    const _Float16* __restrict__ Q, const _Float16* __restrict__ K,
    const _Float16* __restrict__ Vt, _Float16* __restrict__ ctx) {
  __shared__ __align__(16) _Float16 plds[8][16 * 32];
  int lane = threadIdx.x & 31;
  int wave = threadIdx.x >> 5;
  int w = blockIdx.x * 8 + wave;
  int qt = w & 127;
  int h = (w >> 7) & 15;
  int b = w >> 11;
  int kvh = h >> 2;
  int row16 = lane & 15, hi = lane >> 4;

  const _Float16* qrow =
      Q + ((size_t)(b * SEQ + qt * 16 + row16) * N_HEADS + h) * HEAD_DIM;
  v16h afrag[4];
#pragma unroll
  for (int c = 0; c < 4; ++c)
    afrag[c] = pack16(*(const v8h*)(qrow + c * 32 + hi * 8),
                      *(const v8h*)(qrow + c * 32 + hi * 8 + 16));

  const _Float16* kbase = K + ((size_t)b * SEQ * KV_HEADS + kvh) * HEAD_DIM;
  const _Float16* vbase = Vt + ((size_t)b * KV_HEADS + kvh) * (size_t)HEAD_DIM * SEQ;

  v8f acc[8] = {};
  float mrow[8], lrow[8];
#pragma unroll
  for (int r = 0; r < 8; ++r) { mrow[r] = -1e30f; lrow[r] = 0.f; }

  const float scale = 0.08838834764831845f;  // 1/sqrt(128)
  int nkb = (qt * 16 + 16 + 31) >> 5;
  _Float16* pl = plds[wave];

  for (int kb = 0; kb < nkb; ++kb) {
    int k0 = kb * 32;
    v8f sc[2] = {};
#pragma unroll
    for (int sub = 0; sub < 2; ++sub) {
      const _Float16* krow = kbase + (size_t)(k0 + sub * 16 + row16) * KV_DIM;
      // stage the 4 K fragments (pinned), then 4 back-to-back WMMAs
      v16h kf[4];
#pragma unroll
      for (int c = 0; c < 4; ++c)
        kf[c] = pack16(*(const v8h*)(krow + c * 32 + hi * 8),
                       *(const v8h*)(krow + c * 32 + hi * 8 + 16));
      pin4(kf[0], kf[1], kf[2], kf[3]);
#pragma unroll
      for (int c = 0; c < 4; ++c) sc[sub] = wmma16(afrag[c], kf[c], sc[sub]);
    }
    // scale + causal mask (C layout: row = r + hi*8, col = row16)
#pragma unroll
    for (int sub = 0; sub < 2; ++sub) {
      int kpos = k0 + sub * 16 + row16;
#pragma unroll
      for (int r = 0; r < 8; ++r) {
        int qpos = qt * 16 + r + hi * 8;
        float v = sc[sub][r] * scale;
        sc[sub][r] = (kpos <= qpos) ? v : -1e30f;
      }
    }
    // online softmax: row reductions across the 16-lane half
#pragma unroll
    for (int r = 0; r < 8; ++r) {
      float v = fmaxf(sc[0][r], sc[1][r]);
      v = fmaxf(v, __shfl_xor(v, 1, 32));
      v = fmaxf(v, __shfl_xor(v, 2, 32));
      v = fmaxf(v, __shfl_xor(v, 4, 32));
      v = fmaxf(v, __shfl_xor(v, 8, 32));
      float mi = fmaxf(mrow[r], v);
      float alpha = __expf(mrow[r] - mi);
      mrow[r] = mi;
      lrow[r] *= alpha;
#pragma unroll
      for (int j = 0; j < 8; ++j) acc[j][r] *= alpha;
    }
#pragma unroll
    for (int sub = 0; sub < 2; ++sub)
#pragma unroll
      for (int r = 0; r < 8; ++r) sc[sub][r] = __expf(sc[sub][r] - mrow[r]);
#pragma unroll
    for (int r = 0; r < 8; ++r) {
      float v = sc[0][r] + sc[1][r];
      v += __shfl_xor(v, 1, 32);
      v += __shfl_xor(v, 2, 32);
      v += __shfl_xor(v, 4, 32);
      v += __shfl_xor(v, 8, 32);
      lrow[r] += v;
    }
    // C-layout -> A-layout for P via per-wave LDS tile (16x32 f16)
#pragma unroll
    for (int sub = 0; sub < 2; ++sub)
#pragma unroll
      for (int r = 0; r < 8; ++r)
        pl[(r + hi * 8) * 32 + sub * 16 + row16] = (_Float16)sc[sub][r];
    asm volatile("s_wait_dscnt 0" ::: "memory");
    v16h pa = pack16(*(const v8h*)(pl + row16 * 32 + hi * 8),
                     *(const v8h*)(pl + row16 * 32 + hi * 8 + 16));
    // P(16x32) x V(32x128): B column n = d, contiguous along s in Vt
    // two groups of 4: stage 4 V fragments (pinned), then 4 WMMAs
#pragma unroll
    for (int g = 0; g < 2; ++g) {
      v16h vf[4];
#pragma unroll
      for (int dj = 0; dj < 4; ++dj) {
        const _Float16* vcol =
            vbase + (size_t)((g * 4 + dj) * 16 + row16) * SEQ + k0 + hi * 8;
        vf[dj] = pack16(*(const v8h*)vcol, *(const v8h*)(vcol + 16));
      }
      pin4(vf[0], vf[1], vf[2], vf[3]);
#pragma unroll
      for (int dj = 0; dj < 4; ++dj)
        acc[g * 4 + dj] = wmma16(pa, vf[dj], acc[g * 4 + dj]);
    }
  }
  // finalize: ctx[b][m][h][d] = acc / l
#pragma unroll
  for (int dj = 0; dj < 8; ++dj) {
#pragma unroll
    for (int r = 0; r < 8; ++r) {
      int m = qt * 16 + r + hi * 8;
      int d = dj * 16 + row16;
      ctx[((size_t)(b * SEQ + m) * N_HEADS + h) * HEAD_DIM + d] =
          (_Float16)(acc[dj][r] / lrow[r]);
    }
  }
}

// ---------------------------------------------------------------- launch
extern "C" void kernel_launch(void* const* d_in, const int* in_sizes, int n_in,
                              void* d_out, int out_size, void* d_ws, size_t ws_size,
                              hipStream_t stream) {
  (void)in_sizes; (void)n_in; (void)out_size; (void)ws_size;
  const float* x  = (const float*)d_in[0];
  const float* wq = (const float*)d_in[1];
  const float* bq = (const float*)d_in[2];
  const float* wk = (const float*)d_in[3];
  const float* bk = (const float*)d_in[4];
  const float* wv = (const float*)d_in[5];
  const float* bv = (const float*)d_in[6];
  const float* wo = (const float*)d_in[7];
  const float* bo = (const float*)d_in[8];
  const float* cs = (const float*)d_in[9];
  const float* sn = (const float*)d_in[10];

  char* ws = (char*)d_ws;
  size_t off = 0;
  auto alloc = [&](size_t bytes) {
    char* p = ws + off;
    off += (bytes + 255) & ~(size_t)255;
    return p;
  };
  _Float16* xh  = (_Float16*)alloc((size_t)BATCH * SEQ * HID * 2);
  _Float16* wqh = (_Float16*)alloc((size_t)HID * HID * 2);
  _Float16* wkh = (_Float16*)alloc((size_t)KV_DIM * HID * 2);
  _Float16* wvh = (_Float16*)alloc((size_t)KV_DIM * HID * 2);
  _Float16* woh = (_Float16*)alloc((size_t)HID * HID * 2);
  _Float16* qh  = (_Float16*)alloc((size_t)BATCH * SEQ * HID * 2);
  _Float16* kh  = (_Float16*)alloc((size_t)BATCH * SEQ * KV_DIM * 2);
  _Float16* vh  = (_Float16*)alloc((size_t)BATCH * SEQ * KV_DIM * 2);
  _Float16* vth = (_Float16*)alloc((size_t)BATCH * SEQ * KV_DIM * 2);
  _Float16* ctx = (_Float16*)alloc((size_t)BATCH * SEQ * HID * 2);

  auto cvt = [&](const float* src, _Float16* dst, size_t n) {
    hipLaunchKernelGGL(cvt_f32_f16, dim3((unsigned)((n + 255) / 256)), dim3(256),
                       0, stream, src, dst, n);
  };
  cvt(x,  xh,  (size_t)BATCH * SEQ * HID);
  cvt(wq, wqh, (size_t)HID * HID);
  cvt(wk, wkh, (size_t)KV_DIM * HID);
  cvt(wv, wvh, (size_t)KV_DIM * HID);
  cvt(wo, woh, (size_t)HID * HID);

  const int M = BATCH * SEQ;
  auto gemm = [&](const _Float16* A, const _Float16* W, const float* bias,
                  _Float16* oh, float* of, int N, int Kd) {
    int blocks = (M / 128) * (N / 64);
    hipLaunchKernelGGL(gemm_wmma, dim3(blocks), dim3(256), 0, stream,
                       A, W, bias, oh, of, M, N, Kd);
  };
  gemm(xh, wqh, bq, qh, nullptr, HID, HID);
  gemm(xh, wkh, bk, kh, nullptr, KV_DIM, HID);
  gemm(xh, wvh, bv, vh, nullptr, KV_DIM, HID);

  {
    size_t total = (size_t)BATCH * SEQ * (N_HEADS + KV_HEADS) * 64;
    hipLaunchKernelGGL(rope_qk, dim3((unsigned)((total + 255) / 256)), dim3(256),
                       0, stream, qh, kh, cs, sn);
  }
  {
    size_t total = (size_t)BATCH * SEQ * KV_DIM;
    hipLaunchKernelGGL(v_transpose, dim3((unsigned)((total + 255) / 256)),
                       dim3(256), 0, stream, vh, vth);
  }
  {
    int waves = BATCH * N_HEADS * (SEQ / 16);  // 4096 waves, 8 per block
    hipLaunchKernelGGL(flash_attn, dim3(waves / 8), dim3(256), 0, stream,
                       qh, kh, vth, ctx);
  }
  gemm(ctx, woh, bo, nullptr, (float*)d_out, HID, HID);
}
